// MultiHeadAttention_69733089018036
// MI455X (gfx1250) — matrix operand order repaired and checked by reading, not compiled, not run on hardware
//
#include <hip/hip_runtime.h>
#include <hip/hip_bf16.h>

// MI455X / gfx1250 multi-head attention:
//   q = X Wq^T + bq ; k = ... ; v = ...   (f32 in -> f16 out, WMMA f16 GEMM)
//   flash attention per (batch, head)      (WMMA f16, f32 accum, online softmax)
//   out = attn Wo^T + bo                   (f16 in -> f32 out)
// All matrix math uses v_wmma_f32_16x16x32_f16.

#define D_MODEL   1024
#define SEQ       2048
#define NBATCH    4
#define NHEADS    16
#define HEAD_DIM  64
#define MTOK      (NBATCH * SEQ)   // 8192

typedef __attribute__((ext_vector_type(16))) _Float16 v16h;
typedef __attribute__((ext_vector_type(8)))  _Float16 v8h;
typedef __attribute__((ext_vector_type(8)))  float    v8f;

__device__ __forceinline__ v8f wmma_f16(v16h a, v16h b, v8f c) {
  return __builtin_amdgcn_wmma_f32_16x16x32_f16(
      /*neg_a=*/false, a, /*neg_b=*/false, b,
      /*c_mod=*/(short)0, c, /*reuse_a=*/false, /*reuse_b=*/false);
}

// A-matrix fragment (16x32 f16, MxK). Per ISA layout:
//   lane L: m = L%16 ; kb2 = (L<16)?0:8
//   slots 0..7  -> k = kb2 + 0..7      (contiguous)
//   slots 8..15 -> k = kb2 + 16..23    (contiguous)
__device__ __forceinline__ v16h ld_frag_a(const _Float16* rowp, int kb2) {
  v8h lo = *(const v8h*)(rowp + kb2);
  v8h hi = *(const v8h*)(rowp + kb2 + 16);
  v16h r;
#pragma unroll
  for (int i = 0; i < 8; ++i) { r[i] = lo[i]; r[8 + i] = hi[i]; }
  return r;
}

// B-matrix fragment (32x16 f16, KxN). Per ISA layout:
//   lane L: n = L%16 ; kbB = (L<16)?0:8
//   slots 0..7 -> k = kbB + 0..7 ; slots 8..15 -> k = kbB + 16..23
__device__ __forceinline__ v16h ld_frag_b(const _Float16* p) {
  v8h lo = *(const v8h*)(p);
  v8h hi = *(const v8h*)(p + 16);
  v16h r;
#pragma unroll
  for (int i = 0; i < 8; ++i) { r[i] = lo[i]; r[8 + i] = hi[i]; }
  return r;
}

// ---------------------------------------------------------------------------
// GEMM: Y[M, Nout] = X[M, K] @ W^T + bias     (W is [Nout, K] row-major)
// Block: 256 threads (8 waves), 128x64 output tile, k-chunk = 64.
// Wave w (4x2 grid): 32x32 sub-tile -> 4 WMMAs per 32-wide dc chunk,
// 8 WMMAs between barrier pairs; fragment loads reused 2x each.
// ---------------------------------------------------------------------------
template <typename Tin, typename Tout>
__global__ __launch_bounds__(256)
void gemm_xwt(const Tin* __restrict__ X, const float* __restrict__ W,
              const float* __restrict__ bias, Tout* __restrict__ Y,
              int M, int K, int Nout) {
  __shared__ _Float16 Al[128 * 64];   // [m][k]   16 KB
  __shared__ _Float16 Bl[64 * 64];    // [n][k]    8 KB

  const int tid  = threadIdx.x;
  const int lane = tid & 31;
  const int wid  = tid >> 5;
  const int m0 = blockIdx.x * 128;
  const int n0 = blockIdx.y * 64;
  const int wm = (wid & 3) * 32;      // wave m offset (0,32,64,96)
  const int wn = (wid >> 2) * 32;     // wave n offset (0,32)

  // tile-fill indices
  const int ar = tid >> 1;            // 0..127 (A row)
  const int ak = (tid & 1) * 32;      // 0,32
  const int br = tid >> 2;            // 0..63  (B row = out col)
  const int bk = (tid & 3) * 16;      // 0,16,32,48

  const int lidx = lane & 15;
  const int kb2  = (lane < 16) ? 0 : 8;
  const int kbB  = (lane < 16) ? 0 : 8;

  v8f c00 = {}, c01 = {}, c10 = {}, c11 = {};

  for (int k0 = 0; k0 < K; k0 += 64) {
    __syncthreads();
    {
      const Tin* src = X + (size_t)(m0 + ar) * K + k0 + ak;
      _Float16*  dst = Al + ar * 64 + ak;
#pragma unroll
      for (int i = 0; i < 32; ++i) dst[i] = (_Float16)src[i];
      if (k0 + 64 < K)
        __builtin_prefetch(src + 64, 0, 3);     // global_prefetch_b8
    }
    {
      const float* src = W + (size_t)(n0 + br) * K + k0 + bk;
      _Float16*    dst = Bl + br * 64 + bk;
#pragma unroll
      for (int i = 0; i < 16; ++i) dst[i] = (_Float16)src[i];
      if (k0 + 64 < K)
        __builtin_prefetch(src + 64, 0, 3);
    }
    __syncthreads();

#pragma unroll
    for (int dc = 0; dc < 64; dc += 32) {
      v16h a0 = ld_frag_a(Al + (wm + lidx) * 64 + dc, kb2);
      v16h a1 = ld_frag_a(Al + (wm + 16 + lidx) * 64 + dc, kb2);
      v16h b0 = ld_frag_b(Bl + (wn + lidx) * 64 + dc + kbB);
      v16h b1 = ld_frag_b(Bl + (wn + 16 + lidx) * 64 + dc + kbB);
      c00 = wmma_f16(a0, b0, c00);
      c01 = wmma_f16(a0, b1, c01);
      c10 = wmma_f16(a1, b0, c10);
      c11 = wmma_f16(a1, b1, c11);
    }
  }

  // C layout: lane L holds col n=L%16 (+16*frag); VGPR r holds row r (+8 if L>=16)
  const int nc0 = n0 + wn + lidx;
  const int nc1 = nc0 + 16;
  const float bb0 = bias[nc0];
  const float bb1 = bias[nc1];
  const int mb0 = m0 + wm + ((lane >> 4) ? 8 : 0);
  const int mb1 = mb0 + 16;
#pragma unroll
  for (int r = 0; r < 8; ++r) {
    Y[(size_t)(mb0 + r) * Nout + nc0] = (Tout)(c00[r] + bb0);
    Y[(size_t)(mb0 + r) * Nout + nc1] = (Tout)(c01[r] + bb1);
    Y[(size_t)(mb1 + r) * Nout + nc0] = (Tout)(c10[r] + bb0);
    Y[(size_t)(mb1 + r) * Nout + nc1] = (Tout)(c11[r] + bb1);
  }
}

// ---------------------------------------------------------------------------
// Flash attention. Grid: (SEQ/64, NBATCH*NHEADS). Block: 128 threads (4 waves).
// Each wave owns 16 query rows; KV consumed in blocks of 32 (no S x S tensor).
// Buffers are f16, layout [N, S, D_MODEL] with head h at cols [h*64, h*64+64).
// ---------------------------------------------------------------------------
__global__ __launch_bounds__(128)
void flash_attn(const _Float16* __restrict__ Qb, const _Float16* __restrict__ Kb,
                const _Float16* __restrict__ Vb, _Float16* __restrict__ Ob) {
  __shared__ _Float16 Qs[64 * 64];       // [qm][d]
  __shared__ _Float16 Ks[32 * 64];       // [kv][d]
  __shared__ _Float16 Vt[64 * 32];       // [d][kv]  (transposed for B frags)
  __shared__ _Float16 Ps[4][16 * 32];    // per-wave P tile [m][kv]

  const int tid  = threadIdx.x;
  const int lane = tid & 31;
  const int wid  = tid >> 5;
  const int nb = blockIdx.y >> 4;
  const int h  = blockIdx.y & 15;
  const int q0 = blockIdx.x * 64;

  const size_t baseq = ((size_t)nb * SEQ + q0) * D_MODEL + h * HEAD_DIM;
  const size_t basek = (size_t)nb * SEQ * D_MODEL + h * HEAD_DIM;

  // stage Q tile once: 64x64 halves = 512 x v8h
  for (int i = tid; i < 512; i += 128) {
    int row = i >> 3, c8 = (i & 7) * 8;
    *(v8h*)(Qs + row * 64 + c8) =
        *(const v8h*)(Qb + baseq + (size_t)row * D_MODEL + c8);
  }

  const int lidx = lane & 15;
  const int hi16 = lane >> 4;               // 0 or 1
  const int kb2  = hi16 ? 8 : 0;
  const int kbB  = hi16 ? 8 : 0;
  const int qm   = wid * 16 + lidx;

  float Mrow[8], Lrow[8];
#pragma unroll
  for (int r = 0; r < 8; ++r) { Mrow[r] = -1e30f; Lrow[r] = 0.0f; }
  v8f o0 = {}, o1 = {}, o2 = {}, o3 = {};

  for (int kv0 = 0; kv0 < SEQ; kv0 += 32) {
    __syncthreads();
    // K tile: 32x64 = 256 x v8h
    for (int i = tid; i < 256; i += 128) {
      int row = i >> 3, c8 = (i & 7) * 8;
      *(v8h*)(Ks + row * 64 + c8) =
          *(const v8h*)(Kb + basek + (size_t)(kv0 + row) * D_MODEL + c8);
    }
    // V tile, transposed into Vt[d][kv]
    for (int i = tid; i < 256; i += 128) {
      int row = i >> 3, c8 = (i & 7) * 8;
      v8h v = *(const v8h*)(Vb + basek + (size_t)(kv0 + row) * D_MODEL + c8);
#pragma unroll
      for (int j = 0; j < 8; ++j) Vt[(c8 + j) * 32 + row] = v[j];
    }
    __syncthreads();

    // S(16x32) = Q(16x64) @ K^T(64x32): B element [d][kv] = Ks[kv*64+d]
    v8f s0 = {}, s1 = {};
#pragma unroll
    for (int dc = 0; dc < 64; dc += 32) {
      v16h a  = ld_frag_a(Qs + qm * 64 + dc, kb2);
      v16h b0 = ld_frag_b(Ks + lidx * 64 + dc + kbB);
      v16h b1 = ld_frag_b(Ks + (16 + lidx) * 64 + dc + kbB);
      s0 = wmma_f16(a, b0, s0);
      s1 = wmma_f16(a, b1, s1);
    }

    // Online softmax. Row m = r + hi16*8 lives across one 16-lane half at VGPR r.
    float alpha[8];
#pragma unroll
    for (int r = 0; r < 8; ++r) {
      float mx = fmaxf(s0[r], s1[r]);
      mx = fmaxf(mx, __shfl_xor(mx, 1, 32));
      mx = fmaxf(mx, __shfl_xor(mx, 2, 32));
      mx = fmaxf(mx, __shfl_xor(mx, 4, 32));
      mx = fmaxf(mx, __shfl_xor(mx, 8, 32));
      float mn = fmaxf(Mrow[r], mx);
      float al = __expf(Mrow[r] - mn);
      Mrow[r] = mn;
      float p0 = __expf(s0[r] - mn);
      float p1 = __expf(s1[r] - mn);
      float rs = p0 + p1;
      rs += __shfl_xor(rs, 1, 32);
      rs += __shfl_xor(rs, 2, 32);
      rs += __shfl_xor(rs, 4, 32);
      rs += __shfl_xor(rs, 8, 32);
      Lrow[r] = Lrow[r] * al + rs;
      alpha[r] = al;
      // scatter P into per-wave LDS tile [m][kv] for A-fragment reload
      Ps[wid][(r + hi16 * 8) * 32 + lidx]      = (_Float16)p0;
      Ps[wid][(r + hi16 * 8) * 32 + lidx + 16] = (_Float16)p1;
    }
#pragma unroll
    for (int r = 0; r < 8; ++r) {
      o0[r] *= alpha[r]; o1[r] *= alpha[r];
      o2[r] *= alpha[r]; o3[r] *= alpha[r];
    }
    __builtin_amdgcn_wave_barrier();   // per-wave tile; DS ops in-order per wave

    // O(16x64) += P(16x32) @ V(32x64)
    v16h aP = ld_frag_a(Ps[wid] + lidx * 32, kb2);
    {
      v16h bv0 = ld_frag_b(Vt + (0 * 16 + lidx) * 32 + kbB);
      v16h bv1 = ld_frag_b(Vt + (1 * 16 + lidx) * 32 + kbB);
      v16h bv2 = ld_frag_b(Vt + (2 * 16 + lidx) * 32 + kbB);
      v16h bv3 = ld_frag_b(Vt + (3 * 16 + lidx) * 32 + kbB);
      o0 = wmma_f16(aP, bv0, o0);
      o1 = wmma_f16(aP, bv1, o1);
      o2 = wmma_f16(aP, bv2, o2);
      o3 = wmma_f16(aP, bv3, o3);
    }
  }

  // normalize and store (f16, same [N,S,D] layout)
#pragma unroll
  for (int r = 0; r < 8; ++r) {
    float inv = 1.0f / Lrow[r];
    int mrow = wid * 16 + r + hi16 * 8;
    size_t rowoff = baseq + (size_t)mrow * D_MODEL;
    Ob[rowoff + 0 * 16 + lidx] = (_Float16)(o0[r] * inv);
    Ob[rowoff + 1 * 16 + lidx] = (_Float16)(o1[r] * inv);
    Ob[rowoff + 2 * 16 + lidx] = (_Float16)(o2[r] * inv);
    Ob[rowoff + 3 * 16 + lidx] = (_Float16)(o3[r] * inv);
  }
}

// ---------------------------------------------------------------------------

extern "C" void kernel_launch(void* const* d_in, const int* in_sizes, int n_in,
                              void* d_out, int out_size, void* d_ws, size_t ws_size,
                              hipStream_t stream) {
  const float* query = (const float*)d_in[0];
  const float* key   = (const float*)d_in[1];
  const float* value = (const float*)d_in[2];
  const float* Wq = (const float*)d_in[3];
  const float* bq = (const float*)d_in[4];
  const float* Wk = (const float*)d_in[5];
  const float* bk = (const float*)d_in[6];
  const float* Wv = (const float*)d_in[7];
  const float* bv = (const float*)d_in[8];
  const float* Wo = (const float*)d_in[9];
  const float* bo = (const float*)d_in[10];
  float* out = (float*)d_out;

  // workspace: 4 f16 activation buffers of [8192, 1024]  (64 MB total)
  _Float16* qh = (_Float16*)d_ws;
  _Float16* kh = qh + (size_t)MTOK * D_MODEL;
  _Float16* vh = kh + (size_t)MTOK * D_MODEL;
  _Float16* ah = vh + (size_t)MTOK * D_MODEL;

  dim3 gg(MTOK / 128, D_MODEL / 64);   // 64 x 16

  gemm_xwt<float, _Float16><<<gg, 256, 0, stream>>>(query, Wq, bq, qh,
                                                    MTOK, D_MODEL, D_MODEL);
  gemm_xwt<float, _Float16><<<gg, 256, 0, stream>>>(key, Wk, bk, kh,
                                                    MTOK, D_MODEL, D_MODEL);
  gemm_xwt<float, _Float16><<<gg, 256, 0, stream>>>(value, Wv, bv, vh,
                                                    MTOK, D_MODEL, D_MODEL);

  flash_attn<<<dim3(SEQ / 64, NBATCH * NHEADS), 128, 0, stream>>>(qh, kh, vh, ah);

  gemm_xwt<_Float16, float><<<gg, 256, 0, stream>>>(ah, Wo, bo, out,
                                                    MTOK, D_MODEL, D_MODEL);
}
